// VectorizedMoE_83932250899018
// MI455X (gfx1250) — matrix operand order repaired and checked by reading, not compiled, not run on hardware
//
#include <hip/hip_runtime.h>
#include <hip/hip_bf16.h>

// ---------------------------------------------------------------------------
// MoE: B=16384 tokens, H=1024, E=8 experts, K=2 (top-2 routing)
// gate -> per-expert token lists -> grouped bf16 WMMA GEMM (fp32 accum,
// double-buffered async global->LDS copies) -> weighted atomic scatter.
// ---------------------------------------------------------------------------

#define BTOK 16384
#define HDIM 1024
#define NEXP 8

typedef __attribute__((ext_vector_type(16))) __bf16 v16bf;
typedef __attribute__((ext_vector_type(8)))  float  v8f;

// ---- CDNA5 async global->LDS (ASYNCcnt) with safe fallback ----------------
#if defined(__has_builtin)
#if __has_builtin(__builtin_amdgcn_global_load_async_to_lds_b128)
#define HAVE_ASYNC_LDS 1
#endif
#if __has_builtin(__builtin_amdgcn_s_wait_asynccnt)
#define HAVE_WAIT_ASYNC 1
#endif
#endif

#ifdef HAVE_ASYNC_LDS
typedef int v4i_ld __attribute__((vector_size(16)));
typedef __attribute__((address_space(1))) v4i_ld* as1_v4i;
typedef __attribute__((address_space(3))) v4i_ld* as3_v4i;
#endif

static __device__ __forceinline__ void cp_async_b128(const unsigned short* g,
                                                     unsigned short* l) {
#ifdef HAVE_ASYNC_LDS
    __builtin_amdgcn_global_load_async_to_lds_b128(
        (as1_v4i)(g), (as3_v4i)(l), 0, 0);
#else
    *(uint4*)l = *(const uint4*)g;
#endif
}

static __device__ __forceinline__ void wait_async0() {
#ifdef HAVE_ASYNC_LDS
#ifdef HAVE_WAIT_ASYNC
    __builtin_amdgcn_s_wait_asynccnt(0);
#else
    asm volatile("s_wait_asynccnt 0x0" ::: "memory");
#endif
#endif
}

// round-to-nearest-even float -> bf16 bits
static __device__ __forceinline__ unsigned short f2bf(float f) {
    unsigned u = __float_as_uint(f);
    unsigned r = u + 0x7FFFu + ((u >> 16) & 1u);
    return (unsigned short)(r >> 16);
}

// ---------------------------------------------------------------------------
// Kernel 1: convert x (fp32) -> bf16
// ---------------------------------------------------------------------------
__global__ __launch_bounds__(256)
void convert_x_kernel(const float* __restrict__ x, unsigned short* __restrict__ xb) {
    size_t i = ((size_t)blockIdx.x * 256 + threadIdx.x) * 4;
    float4 v = *(const float4*)(x + i);
    ushort4 o;
    o.x = f2bf(v.x); o.y = f2bf(v.y); o.z = f2bf(v.z); o.w = f2bf(v.w);
    *(ushort4*)(xb + i) = o;
}

// ---------------------------------------------------------------------------
// Kernel 2: WeT[e][d][h] = bf16( We[e][h][d] )  (tiled transpose via LDS)
// grid (H/32, H/32, E), block (32, 8)
// ---------------------------------------------------------------------------
__global__ __launch_bounds__(256)
void transpose_we_kernel(const float* __restrict__ We, unsigned short* __restrict__ wet) {
    __shared__ float tile[32][33];
    int e  = blockIdx.z;
    int d0 = blockIdx.x * 32;
    int h0 = blockIdx.y * 32;
    const float* in = We + (size_t)e * HDIM * HDIM;
    unsigned short* out = wet + (size_t)e * HDIM * HDIM;
#pragma unroll
    for (int j = 0; j < 4; ++j)
        tile[threadIdx.y + j * 8][threadIdx.x] =
            in[(size_t)(h0 + threadIdx.y + j * 8) * HDIM + d0 + threadIdx.x];
    __syncthreads();
#pragma unroll
    for (int j = 0; j < 4; ++j)
        out[(size_t)(d0 + threadIdx.y + j * 8) * HDIM + h0 + threadIdx.x] =
            f2bf(tile[threadIdx.x][threadIdx.y + j * 8]);
}

// ---------------------------------------------------------------------------
// Kernel 3: gating. One wave per token: logits = x@Wg + bg (wave32 shfl
// reduction), top-2, softmax over 2, atomic append (token, weight).
// ---------------------------------------------------------------------------
__global__ __launch_bounds__(256)
void gate_kernel(const float* __restrict__ x, const float* __restrict__ Wg,
                 const float* __restrict__ bg, int* __restrict__ counts,
                 int* __restrict__ lists, float* __restrict__ wlist) {
    int wave = threadIdx.x >> 5;
    int lane = threadIdx.x & 31;
    int b = blockIdx.x * 8 + wave;

    const float* xr = x + (size_t)b * HDIM;
    float acc[NEXP];
#pragma unroll
    for (int e = 0; e < NEXP; ++e) acc[e] = 0.f;

    for (int h = lane; h < HDIM; h += 32) {
        float xv = xr[h];
        const float* wr = Wg + (size_t)h * NEXP;
#pragma unroll
        for (int e = 0; e < NEXP; ++e) acc[e] = fmaf(xv, wr[e], acc[e]);
    }
#pragma unroll
    for (int e = 0; e < NEXP; ++e) {
        float v = acc[e];
#pragma unroll
        for (int m = 16; m > 0; m >>= 1) v += __shfl_xor(v, m, 32);
        acc[e] = v + bg[e];
    }
    if (lane == 0) {
        int i0 = 0; float v0 = acc[0];
#pragma unroll
        for (int e = 1; e < NEXP; ++e)
            if (acc[e] > v0) { v0 = acc[e]; i0 = e; }
        int i1 = -1; float v1 = -3.4e38f;
#pragma unroll
        for (int e = 0; e < NEXP; ++e)
            if (e != i0 && acc[e] > v1) { v1 = acc[e]; i1 = e; }
        float p  = __expf(v1 - v0);       // v1 <= v0 so p in (0,1]
        float w0 = 1.f / (1.f + p);
        float w1 = p * w0;
        int p0 = atomicAdd(&counts[i0], 1);
        lists[i0 * BTOK + p0] = b; wlist[i0 * BTOK + p0] = w0;
        int p1 = atomicAdd(&counts[i1], 1);
        lists[i1 * BTOK + p1] = b; wlist[i1 * BTOK + p1] = w1;
    }
}

// ---------------------------------------------------------------------------
// Kernel 4: grouped GEMM per expert, v_wmma_f32_16x16x32_bf16.
// grid (H/128, B/128, E), block 256 (8 waves).
// Tile M=128 x N=128, K-step 64, double-buffered async global->LDS.
// Wave w -> rows [(w&3)*32, +32), cols [(w>>2)*64, +64).
// Padding rows are clamped to a valid token (dead accumulator rows are
// simply never stored) so the hot loop has no divergence.
// ---------------------------------------------------------------------------
#define TM 128
#define TN 128
#define KS 64
#define LDK 72   // ushort stride: 64 data + 8 pad => 144 B rows (16B aligned)

__global__ __launch_bounds__(256)
void moe_gemm_kernel(const unsigned short* __restrict__ xb,
                     const unsigned short* __restrict__ wet,
                     const int* __restrict__ counts,
                     const int* __restrict__ lists,
                     const float* __restrict__ wlist,
                     const float* __restrict__ be,
                     float* __restrict__ out) {
    int e     = blockIdx.z;
    int mBase = blockIdx.y * TM;
    int nBase = blockIdx.x * TN;
    int cnt   = counts[e];
    if (mBase >= cnt) return;
    int rowsValid = cnt - mBase;
    if (rowsValid > TM) rowsValid = TM;

    __shared__ __align__(16) unsigned short As[2][TM * LDK];   // 2 x 18 KB
    __shared__ __align__(16) unsigned short Bs[2][TN * LDK];   // 2 x 18 KB

    int tid  = threadIdx.x;
    int wave = tid >> 5, lane = tid & 31;
    int wm = wave & 3, wn = wave >> 2;
    int lMod = lane & 15, lHalf = lane >> 4;

    // Staging: tile = 128 rows x 64 bf16 = 1024 chunks of 8 bf16 (16 B).
    // Thread owns chunks at rows (tid>>3)+32*s, quad q=(tid&7), s=0..3.
    int rBase = tid >> 3;
    int q     = tid & 7;
    const unsigned short* wbase = wet + (size_t)e * HDIM * HDIM;

    long tokA[4];
    const unsigned short* bsrc[4];
#pragma unroll
    for (int s = 0; s < 4; ++s) {
        int r = rBase + 32 * s;
        int lr = (r < rowsValid) ? r : (rowsValid - 1);   // clamp padding rows
        tokA[s] = (long)lists[e * BTOK + mBase + lr];
        bsrc[s] = wbase + (size_t)(nBase + r) * HDIM + q * 8;
    }

    v8f acc[2][4];
#pragma unroll
    for (int i = 0; i < 2; ++i)
#pragma unroll
        for (int j = 0; j < 4; ++j) acc[i][j] = {};

    // ---- prologue: stage k0 = 0 into buffer 0 ----
#pragma unroll
    for (int s = 0; s < 4; ++s) {
        int r = rBase + 32 * s;
        cp_async_b128(xb + (size_t)tokA[s] * HDIM + q * 8, &As[0][r * LDK + q * 8]);
        cp_async_b128(bsrc[s], &Bs[0][r * LDK + q * 8]);
    }
    wait_async0();
    __syncthreads();

    const int NSTAGE = HDIM / KS;   // 16
    for (int st = 0; st < NSTAGE; ++st) {
        int cur = st & 1;
        // ---- issue next stage's async copies into the other buffer ----
        if (st + 1 < NSTAGE) {
            int nxt = cur ^ 1;
            int k0n = (st + 1) * KS;
#pragma unroll
            for (int s = 0; s < 4; ++s) {
                int r = rBase + 32 * s;
                cp_async_b128(xb + (size_t)tokA[s] * HDIM + k0n + q * 8,
                              &As[nxt][r * LDK + q * 8]);
                cp_async_b128(bsrc[s] + k0n, &Bs[nxt][r * LDK + q * 8]);
            }
        }

        // ---- fragments per ISA 16-bit layouts (CDNA5 ISA §7.12.2) ----
        // A 16x32: lane half0 K=0..7,16..23 ; half1 K=8..15,24..31
        v16bf af[2][2];
#pragma unroll
        for (int i = 0; i < 2; ++i)
#pragma unroll
            for (int kk = 0; kk < 2; ++kk) {
                const unsigned short* p =
                    &As[cur][(wm * 32 + i * 16 + lMod) * LDK + kk * 32 + lHalf * 8];
                ((uint4*)&af[i][kk])[0] = *(const uint4*)(p);
                ((uint4*)&af[i][kk])[1] = *(const uint4*)(p + 16);
            }
        // B 32x16: lane half0 K=0..7,16..23 ; half1 K=8..15,24..31
#pragma unroll
        for (int j = 0; j < 4; ++j) {
            v16bf bfr[2];
#pragma unroll
            for (int kk = 0; kk < 2; ++kk) {
                const unsigned short* p =
                    &Bs[cur][(wn * 64 + j * 16 + lMod) * LDK + kk * 32 + lHalf * 8];
                ((uint4*)&bfr[kk])[0] = *(const uint4*)(p);
                ((uint4*)&bfr[kk])[1] = *(const uint4*)(p + 16);
            }
#pragma unroll
            for (int i = 0; i < 2; ++i) {
                acc[i][j] = __builtin_amdgcn_wmma_f32_16x16x32_bf16(
                    false, af[i][0], false, bfr[0], (short)0, acc[i][j], false, false);
                acc[i][j] = __builtin_amdgcn_wmma_f32_16x16x32_bf16(
                    false, af[i][1], false, bfr[1], (short)0, acc[i][j], false, false);
            }
        }

        wait_async0();       // own async writes for next stage have landed
        __syncthreads();     // everyone done reading cur + everyone's writes done
    }

    // ---- weighted scatter: out[token] += w * (acc + be[e]) ----
#pragma unroll
    for (int i = 0; i < 2; ++i) {
        int  toks[8]; float wts[8];
#pragma unroll
        for (int vg = 0; vg < 8; ++vg) {
            int lr = wm * 32 + i * 16 + lHalf * 8 + vg;   // C/D: VGPR j -> M = j + 8*laneHalf
            if (lr < rowsValid) {
                toks[vg] = lists[e * BTOK + mBase + lr];
                wts[vg]  = wlist[e * BTOK + mBase + lr];
            } else {
                toks[vg] = -1; wts[vg] = 0.f;
            }
        }
#pragma unroll
        for (int j = 0; j < 4; ++j) {
            int n = nBase + wn * 64 + j * 16 + lMod;
            float bev = be[e * HDIM + n];
#pragma unroll
            for (int vg = 0; vg < 8; ++vg) {
                if (toks[vg] >= 0)
                    atomicAdd(out + (size_t)toks[vg] * HDIM + n,
                              wts[vg] * (acc[i][j][vg] + bev));
            }
        }
    }
}

// ---------------------------------------------------------------------------
// Host launcher. Workspace layout (bytes):
//   [0, 32)            : int counts[8]
//   [4096, +512KB)     : int lists[E*B]
//   [528384, +512KB)   : float wlist[E*B]
//   [1052672, +32MB)   : bf16 xb[B*H]
//   [34607104, +16MB)  : bf16 WeT[E*H*H]   (total ~49 MB)
// ---------------------------------------------------------------------------
extern "C" void kernel_launch(void* const* d_in, const int* in_sizes, int n_in,
                              void* d_out, int out_size, void* d_ws, size_t ws_size,
                              hipStream_t stream) {
    const float* x  = (const float*)d_in[0];
    const float* Wg = (const float*)d_in[1];
    const float* bg = (const float*)d_in[2];
    const float* We = (const float*)d_in[3];
    const float* be = (const float*)d_in[4];
    float* out = (float*)d_out;

    char* ws = (char*)d_ws;
    int*            counts = (int*)(ws);
    int*            lists  = (int*)(ws + 4096);
    float*          wlist  = (float*)(ws + 4096 + 524288);
    unsigned short* xb     = (unsigned short*)(ws + 1052672);
    unsigned short* wet    = (unsigned short*)(ws + 34607104);

    (void)hipMemsetAsync(counts, 0, NEXP * sizeof(int), stream);
    (void)hipMemsetAsync(out, 0, (size_t)BTOK * HDIM * sizeof(float), stream);

    convert_x_kernel<<<(BTOK * HDIM) / (256 * 4), 256, 0, stream>>>(x, xb);
    transpose_we_kernel<<<dim3(HDIM / 32, HDIM / 32, NEXP), dim3(32, 8), 0, stream>>>(We, wet);
    gate_kernel<<<BTOK / 8, 256, 0, stream>>>(x, Wg, bg, counts, lists, wlist);
    moe_gemm_kernel<<<dim3(HDIM / TN, BTOK / TM, NEXP), 256, 0, stream>>>(
        xb, wet, counts, lists, wlist, be, out);
}
